// TradeGNN_60705067761929
// MI455X (gfx1250) — matrix operand order repaired and checked by reading, not compiled, not run on hardware
//
#include <hip/hip_runtime.h>
#include <hip/hip_bf16.h>

// ---------------------------------------------------------------------------
// TradeGNN forward for MI455X (gfx1250, wave32, WMMA f16 -> f32,
// async global->LDS staging + prefetch in the GEMM)
// ---------------------------------------------------------------------------

typedef __attribute__((ext_vector_type(16))) _Float16 v16h;
typedef __attribute__((ext_vector_type(8)))  _Float16 v8h;
typedef __attribute__((ext_vector_type(8)))  float    v8f;

#define NN  50000      // nodes
#define NE  800000     // edges
#define ET  (NE + NN)  // edges + self loops
#define HIDC 128

// ---------------------------------------------------------------------------
// Generic helpers
// ---------------------------------------------------------------------------

__global__ void k_fill(float* __restrict__ p, float v, size_t n) {
  size_t id = (size_t)blockIdx.x * blockDim.x + threadIdx.x;
  if (id < n) p[id] = v;
}

// dst[Rpad x C] (f16) = src[R x C] (f32), rows >= R zero-filled
__global__ void k_cast_pad_f16(const float* __restrict__ src, _Float16* __restrict__ dst,
                               int R, int Rpad, int C) {
  size_t id = (size_t)blockIdx.x * blockDim.x + threadIdx.x;
  if (id >= (size_t)Rpad * C) return;
  int r = (int)(id / C);
  int c = (int)(id % C);
  dst[id] = (r < R) ? (_Float16)src[(size_t)r * C + c] : (_Float16)0.0f;
}

// ---------------------------------------------------------------------------
// WMMA f16 GEMM:  C[M,N] (f32) = A[M,K] (f16 row-major) * B[K,N] (f16 row-major)
// K multiple of 32, N multiple of BN. M ragged.
// 256 threads = 8 waves; per-wave tile 64x32 (4x2 WMMA accumulators).
// A tile staged with GLOBAL_LOAD_ASYNC_TO_LDS_B128 (ASYNCcnt) for full tiles.
// ---------------------------------------------------------------------------
template <int BN>
__global__ __launch_bounds__(256) void k_gemm_wmma(
    const _Float16* __restrict__ A, const _Float16* __restrict__ B,
    float* __restrict__ C, int M, int N, int K) {
  constexpr int WN = BN / 32;   // waves along N
  constexpr int WM = 8 / WN;    // waves along M
  constexpr int BM = WM * 64;   // block tile M

  __shared__ __align__(16) _Float16 lA[BM][40];  // [row][k]  (pad to 40 halves)
  __shared__ __align__(16) _Float16 lBt[BN][40]; // [n][k]    (B transposed)

  const int tid  = threadIdx.x;
  const int lane = tid & 31;
  const int wave = tid >> 5;
  const int r16  = lane & 15;
  const int hi   = lane >> 4;           // 0: lanes 0-15, 1: lanes 16-31
  const int wmB  = (wave / WN) * 64;    // wave M offset inside block
  const int wnB  = (wave % WN) * 32;    // wave N offset inside block
  const int m0   = blockIdx.x * BM;
  const int n0   = blockIdx.y * BN;

  const bool fullTile = (m0 + BM <= M);   // uniform across the block

  v8f acc[4][2] = {};

  for (int k0 = 0; k0 < K; k0 += 32) {
    // ---- prefetch next K-slice (global_prefetch_b8) ----
    if (k0 + 32 < K) {
      int pr = m0 + (tid >> 2);
      if (pr < M) __builtin_prefetch(A + (size_t)pr * K + k0 + 32 + (tid & 3) * 8, 0, 0);
      __builtin_prefetch(B + (size_t)(k0 + 32 + (tid >> 3)) * N + n0 + (tid & 7) * 16, 0, 0);
    }

    // ---- stage A tile (BM x 32) ----
    constexpr int ACH = BM * 32 / 8;       // 8-half (16B) chunks
    if (fullTile) {
      // async DMA directly into LDS; per-lane 16B, tracked by ASYNCcnt
      #pragma unroll
      for (int c = 0; c < ACH / 256; ++c) {
        int ch  = c * 256 + tid;
        int row = ch >> 2;                 // 4 chunks per row
        int cc  = ch & 3;
        const _Float16* gp = A + (size_t)(m0 + row) * K + k0 + cc * 8;
        unsigned lds = (unsigned)(unsigned long long)(&lA[row][cc * 8]);
        asm volatile("global_load_async_to_lds_b128 %0, %1, off"
                     :: "v"(lds), "v"(gp) : "memory");
      }
    } else {
      #pragma unroll
      for (int c = 0; c < ACH / 256; ++c) {
        int ch  = c * 256 + tid;
        int row = ch >> 2;
        int cc  = ch & 3;
        int gr  = m0 + row;
        v8h v = {};
        if (gr < M) v = *(const v8h*)(A + (size_t)gr * K + k0 + cc * 8);
        *(v8h*)&lA[row][cc * 8] = v;
      }
    }
    // ---- stage B tile (32 x BN) transposed into lBt[n][k] ----
    constexpr int BCH = 32 * BN / 8;
    #pragma unroll
    for (int c = 0; c < BCH / 256; ++c) {
      int ch = c * 256 + tid;
      int kr = ch / (BN / 8);
      int nc = ch % (BN / 8);
      v8h v = *(const v8h*)(B + (size_t)(k0 + kr) * N + n0 + nc * 8);
      #pragma unroll
      for (int j = 0; j < 8; ++j) lBt[nc * 8 + j][kr] = v[j];
    }
    if (fullTile) asm volatile("s_wait_asynccnt 0x0" ::: "memory");
    __syncthreads();

    // ---- fragments per ISA layout ----
    v16h af[4], bf[2];
    #pragma unroll
    for (int i = 0; i < 4; ++i) {
      int row = wmB + i * 16 + r16;
      v8h lo = *(const v8h*)&lA[row][hi * 8];        // k = hi*8 .. +7
      v8h hh = *(const v8h*)&lA[row][16 + hi * 8];   // k = 16+hi*8 .. +7
      af[i] = __builtin_shufflevector(lo, hh, 0,1,2,3,4,5,6,7,8,9,10,11,12,13,14,15);
    }
    #pragma unroll
    for (int j = 0; j < 2; ++j) {
      int n = wnB + j * 16 + r16;
      v8h lo = *(const v8h*)&lBt[n][hi * 8];         // k = hi*8 .. +7
      v8h hh = *(const v8h*)&lBt[n][16 + hi * 8];    // k = 16+hi*8 .. +7
      bf[j] = __builtin_shufflevector(lo, hh, 0,1,2,3,4,5,6,7,8,9,10,11,12,13,14,15);
    }
    #pragma unroll
    for (int i = 0; i < 4; ++i)
      #pragma unroll
      for (int j = 0; j < 2; ++j)
        acc[i][j] = __builtin_amdgcn_wmma_f32_16x16x32_f16(
            false, af[i], false, bf[j], (short)0, acc[i][j], false, false);
    __syncthreads();
  }

  // ---- store D (C/D layout: m = v + hi*8, n = lane%16) ----
  #pragma unroll
  for (int i = 0; i < 4; ++i)
    #pragma unroll
    for (int j = 0; j < 2; ++j) {
      int col = n0 + wnB + j * 16 + r16;
      #pragma unroll
      for (int v = 0; v < 8; ++v) {
        int row = m0 + wmB + i * 16 + hi * 8 + v;
        if (row < M) C[(size_t)row * N + col] = acc[i][j][v];
      }
    }
}

// ---------------------------------------------------------------------------
// Layer 0 linear (K=4 is too small for WMMA): z[M,N] = x[M,4] @ W[4,N]
// ---------------------------------------------------------------------------
__global__ void k_linear_small(const float* __restrict__ x, const float* __restrict__ W,
                               float* __restrict__ z, int M, int N, int K) {
  size_t id = (size_t)blockIdx.x * blockDim.x + threadIdx.x;
  if (id >= (size_t)M * N) return;
  int n = (int)(id / N), j = (int)(id % N);
  float s = 0.f;
  for (int k = 0; k < K; ++k) s += x[(size_t)n * K + k] * W[(size_t)k * N + j];
  z[id] = s;
}

// ---------------------------------------------------------------------------
// GAT attention pieces
// ---------------------------------------------------------------------------
__global__ void k_attn_logits(const float* __restrict__ z, const float* __restrict__ as,
                              const float* __restrict__ ad, float* __restrict__ als,
                              float* __restrict__ ald, int n, int H, int C) {
  size_t id = (size_t)blockIdx.x * blockDim.x + threadIdx.x;
  if (id >= (size_t)n * H) return;
  int node = (int)(id / H), h = (int)(id % H);
  const float* zp = z + (size_t)node * H * C + (size_t)h * C;
  float s = 0.f, d = 0.f;
  for (int c = 0; c < C; ++c) {
    float v = zp[c];
    s += v * as[h * C + c];
    d += v * ad[h * C + c];
  }
  als[id] = s;
  ald[id] = d;
}

__device__ inline void atomicMaxF(float* addr, float v) {
  if (v >= 0.0f) atomicMax((int*)addr, __float_as_int(v));
  else           atomicMin((unsigned int*)addr, __float_as_uint(v));
}

// pass 1: e = leaky_relu(als[src]+ald[dst]); segment max over dst
__global__ void k_edge_max(const int* __restrict__ ei, const float* __restrict__ als,
                           const float* __restrict__ ald, float* __restrict__ eb,
                           float* __restrict__ m, int H) {
  size_t id = (size_t)blockIdx.x * blockDim.x + threadIdx.x;
  if (id >= (size_t)ET * H) return;
  int i = (int)(id / H), h = (int)(id % H);
  int src = (i < NE) ? ei[i] : (i - NE);
  int dst = (i < NE) ? ei[NE + i] : (i - NE);
  float e = als[(size_t)src * H + h] + ald[(size_t)dst * H + h];
  e = (e > 0.f) ? e : 0.2f * e;                // LeakyReLU(0.2)
  eb[id] = e;
  atomicMaxF(&m[(size_t)dst * H + h], e);
}

// pass 2: w = exp(e - m[dst]); segment sum over dst (w stored back into eb)
__global__ void k_edge_expsum(const int* __restrict__ ei, float* __restrict__ eb,
                              const float* __restrict__ m, float* __restrict__ s, int H) {
  size_t id = (size_t)blockIdx.x * blockDim.x + threadIdx.x;
  if (id >= (size_t)ET * H) return;
  int i = (int)(id / H), h = (int)(id % H);
  int dst = (i < NE) ? ei[NE + i] : (i - NE);
  float w = expf(eb[id] - m[(size_t)dst * H + h]);
  eb[id] = w;
  atomicAdd(&s[(size_t)dst * H + h], w);
}

// pass 3: out[dst,h,c] += z[src,h,c] * w/(s[dst,h]+1e-16)
__global__ void k_edge_scatter(const int* __restrict__ ei, const float* __restrict__ eb,
                               const float* __restrict__ s, const float* __restrict__ z,
                               float* __restrict__ out, int H, int C) {
  size_t id = (size_t)blockIdx.x * blockDim.x + threadIdx.x;
  size_t tot = (size_t)ET * H * C;
  if (id >= tot) return;
  int c = (int)(id % C);
  int h = (int)((id / C) % H);
  int i = (int)(id / ((size_t)C * H));
  int src = (i < NE) ? ei[i] : (i - NE);
  int dst = (i < NE) ? ei[NE + i] : (i - NE);
  float alpha = eb[(size_t)i * H + h] / (s[(size_t)dst * H + h] + 1e-16f);
  atomicAdd(&out[((size_t)dst * H + h) * C + c],
            z[((size_t)src * H + h) * C + c] * alpha);
}

// ---------------------------------------------------------------------------
// BatchNorm (batch statistics) + activation
// ---------------------------------------------------------------------------
__global__ void k_colsum(const float* __restrict__ x, float* __restrict__ sum,
                         float* __restrict__ sq, int rows, int ch) {
  int c = blockIdx.y * blockDim.x + threadIdx.x;
  if (c >= ch) return;
  float s = 0.f, q = 0.f;
  for (int r = blockIdx.x; r < rows; r += gridDim.x) {
    float v = x[(size_t)r * ch + c];
    s += v; q += v * v;
  }
  atomicAdd(&sum[c], s);
  atomicAdd(&sq[c], q);
}

__global__ void k_bn_fin(float* __restrict__ mean, float* __restrict__ istd,
                         int rows, int ch) {
  int c = blockIdx.x * blockDim.x + threadIdx.x;
  if (c >= ch) return;
  float m = mean[c] / (float)rows;
  float v = istd[c] / (float)rows - m * m;
  mean[c] = m;
  istd[c] = rsqrtf(v + 1e-5f);
}

// mode 0: ELU, mode 1: ReLU.  y32/y16 optional outputs (f32 and f16 copies).
__global__ void k_bn_act(const float* __restrict__ x, const float* __restrict__ mean,
                         const float* __restrict__ istd, const float* __restrict__ g,
                         const float* __restrict__ be, float* __restrict__ y32,
                         _Float16* __restrict__ y16, size_t rows, int ch, int mode) {
  size_t id = (size_t)blockIdx.x * blockDim.x + threadIdx.x;
  if (id >= rows * (size_t)ch) return;
  int c = (int)(id % ch);
  float v = (x[id] - mean[c]) * istd[c] * g[c] + be[c];
  v = (mode == 0) ? ((v > 0.f) ? v : (expf(v) - 1.f))   // ELU
                  : fmaxf(v, 0.f);                       // ReLU
  if (y32) y32[id] = v;
  if (y16) y16[id] = (_Float16)v;
}

// y = relu(y + bias)  (MLP layer 2: no BN, bias matters)
__global__ void k_bias_relu(float* __restrict__ y, const float* __restrict__ b,
                            size_t rows, int ch) {
  size_t id = (size_t)blockIdx.x * blockDim.x + threadIdx.x;
  if (id >= rows * (size_t)ch) return;
  y[id] = fmaxf(y[id] + b[id % ch], 0.f);
}

// ---------------------------------------------------------------------------
// emb[E,288] (f16) = [ h[row] (128) | h[col] (128) | edge_attr (10) | 0-pad(22) ]
// ---------------------------------------------------------------------------
__global__ void k_gather_emb(const float* __restrict__ h, const int* __restrict__ ei,
                             const float* __restrict__ ea, _Float16* __restrict__ emb) {
  size_t id = (size_t)blockIdx.x * blockDim.x + threadIdx.x;
  if (id >= (size_t)NE * 288) return;
  int e = (int)(id / 288), c = (int)(id % 288);
  float v = 0.f;
  if (c < 128)       v = h[(size_t)ei[e] * HIDC + c];
  else if (c < 256)  v = h[(size_t)ei[NE + e] * HIDC + (c - 128)];
  else if (c < 266)  v = ea[(size_t)e * 10 + (c - 256)];
  emb[id] = (_Float16)v;
}

// out[e] = y2[e,:64] . mw3 + mb3
__global__ void k_final_dot(const float* __restrict__ y2, const float* __restrict__ w,
                            const float* __restrict__ b, float* __restrict__ out) {
  size_t e = (size_t)blockIdx.x * blockDim.x + threadIdx.x;
  if (e >= NE) return;
  float s = b[0];
  const float* yp = y2 + e * 64;
  for (int i = 0; i < 64; ++i) s += yp[i] * w[i];
  out[e] = s;
}

// ---------------------------------------------------------------------------
// Host orchestration
// ---------------------------------------------------------------------------
extern "C" void kernel_launch(void* const* d_in, const int* in_sizes, int n_in,
                              void* d_out, int out_size, void* d_ws, size_t ws_size,
                              hipStream_t stream) {
  const float* x  = (const float*)d_in[0];
  const int*   ei = (const int*)d_in[1];
  const float* ea = (const float*)d_in[2];
  // params (dict order): per conv layer i: W, as, ad, b, g, be
  const float* W[3]  = {(const float*)d_in[3],  (const float*)d_in[9],  (const float*)d_in[15]};
  const float* AS[3] = {(const float*)d_in[4],  (const float*)d_in[10], (const float*)d_in[16]};
  const float* AD[3] = {(const float*)d_in[5],  (const float*)d_in[11], (const float*)d_in[17]};
  const float* G[3]  = {(const float*)d_in[7],  (const float*)d_in[13], (const float*)d_in[19]};
  const float* BE[3] = {(const float*)d_in[8],  (const float*)d_in[14], (const float*)d_in[20]};
  const float* mw0 = (const float*)d_in[21];
  const float* mw1 = (const float*)d_in[23];
  const float* mw2 = (const float*)d_in[25];
  const float* mb2 = (const float*)d_in[26];
  const float* mw3 = (const float*)d_in[27];
  const float* mb3 = (const float*)d_in[28];
  const float* mg[2]  = {(const float*)d_in[29], (const float*)d_in[31]};
  const float* mbe[2] = {(const float*)d_in[30], (const float*)d_in[32]};
  float* out = (float*)d_out;
  // NOTE: GAT biases b{i} and MLP biases mb0/mb1 cancel inside batch-stat BN.

  // ---- workspace carve ----
  char* w8 = (char*)d_ws;
  size_t off = 0;
  auto take = [&](size_t bytes) { void* p = w8 + off; off += (bytes + 255) & ~(size_t)255; return p; };
  float*    H32  = (float*)   take((size_t)NN * 512 * 4);  // node features / GAT out
  float*    Z32  = (float*)   take((size_t)NN * 512 * 4);  // z (linear output)
  _Float16* A16  = (_Float16*)take((size_t)NN * 512 * 2);  // f16 copy of h for GEMM A
  float*    ALS  = (float*)   take((size_t)NN * 4 * 4);
  float*    ALD  = (float*)   take((size_t)NN * 4 * 4);
  float*    MB   = (float*)   take((size_t)NN * 4 * 4);
  float*    SB   = (float*)   take((size_t)NN * 4 * 4);
  float*    EB   = (float*)   take((size_t)ET * 4 * 4);
  _Float16* W16  = (_Float16*)take((size_t)512 * 512 * 2); // staged f16 weights
  float*    MEAN = (float*)   take(512 * 4);
  float*    ISTD = (float*)   take(512 * 4);
  _Float16* EMB  = (_Float16*)take((size_t)NE * 288 * 2);  // emb / MLP f16 activations
  float*    Y32  = (float*)   take((size_t)NE * 256 * 4);  // MLP f32 GEMM outputs
  (void)ws_size; (void)in_sizes; (void)n_in; (void)out_size;

  auto blk = [](size_t n) { return dim3((unsigned)((n + 255) / 256)); };
  auto fill = [&](float* p, float v, size_t n) { k_fill<<<blk(n), 256, 0, stream>>>(p, v, n); };

  auto bn_stats = [&](const float* xin, size_t rows, int ch) {
    fill(MEAN, 0.f, ch); fill(ISTD, 0.f, ch);
    k_colsum<<<dim3(256, (unsigned)((ch + 255) / 256)), 256, 0, stream>>>(xin, MEAN, ISTD, (int)rows, ch);
    k_bn_fin<<<blk(ch), 256, 0, stream>>>(MEAN, ISTD, (int)rows, ch);
  };

  // attention softmax + scatter + BN + ELU for one GAT layer (z in Z32 -> h in H32)
  auto gat_tail = [&](int l, int H, bool wantF16) {
    int HC = H * HIDC;
    k_attn_logits<<<blk((size_t)NN * H), 256, 0, stream>>>(Z32, AS[l], AD[l], ALS, ALD, NN, H, HIDC);
    fill(MB, -1e30f, (size_t)NN * H);
    fill(SB, 0.f,    (size_t)NN * H);
    k_edge_max<<<blk((size_t)ET * H), 256, 0, stream>>>(ei, ALS, ALD, EB, MB, H);
    k_edge_expsum<<<blk((size_t)ET * H), 256, 0, stream>>>(ei, EB, MB, SB, H);
    fill(H32, 0.f, (size_t)NN * HC);
    k_edge_scatter<<<blk((size_t)ET * HC), 256, 0, stream>>>(ei, EB, SB, Z32, H32, H, HIDC);
    bn_stats(H32, NN, HC);
    k_bn_act<<<blk((size_t)NN * HC), 256, 0, stream>>>(H32, MEAN, ISTD, G[l], BE[l],
                                                       H32, wantF16 ? A16 : nullptr,
                                                       (size_t)NN, HC, /*ELU*/0);
  };

  // ---- GAT layer 0 (K=4: scalar linear) ----
  k_linear_small<<<blk((size_t)NN * 512), 256, 0, stream>>>(x, W[0], Z32, NN, 512, 4);
  gat_tail(0, 4, true);

  // ---- GAT layer 1: [50000,512] x [512,512] ----
  k_cast_pad_f16<<<blk((size_t)512 * 512), 256, 0, stream>>>(W[1], W16, 512, 512, 512);
  k_gemm_wmma<128><<<dim3((NN + 127) / 128, 512 / 128), 256, 0, stream>>>(A16, W16, Z32, NN, 512, 512);
  gat_tail(1, 4, true);

  // ---- GAT layer 2: [50000,512] x [512,128], single head ----
  k_cast_pad_f16<<<blk((size_t)512 * 128), 256, 0, stream>>>(W[2], W16, 512, 512, 128);
  k_gemm_wmma<128><<<dim3((NN + 127) / 128, 128 / 128), 256, 0, stream>>>(A16, W16, Z32, NN, 128, 512);
  gat_tail(2, 1, false);

  // ---- edge MLP ----
  // emb [E,288] f16 (K padded 266 -> 288)
  k_gather_emb<<<blk((size_t)NE * 288), 256, 0, stream>>>(H32, ei, ea, EMB);

  // L0: [E,288] x [288,256] -> Y32 ; BN + ReLU -> f16 in EMB (stride 256)
  k_cast_pad_f16<<<blk((size_t)288 * 256), 256, 0, stream>>>(mw0, W16, 266, 288, 256);
  k_gemm_wmma<128><<<dim3((NE + 127) / 128, 256 / 128), 256, 0, stream>>>(EMB, W16, Y32, NE, 256, 288);
  bn_stats(Y32, NE, 256);
  k_bn_act<<<blk((size_t)NE * 256), 256, 0, stream>>>(Y32, MEAN, ISTD, mg[0], mbe[0],
                                                      nullptr, EMB, (size_t)NE, 256, /*ReLU*/1);

  // L1: [E,256] x [256,128] -> Y32 ; BN + ReLU -> f16 in EMB (stride 128)
  k_cast_pad_f16<<<blk((size_t)256 * 128), 256, 0, stream>>>(mw1, W16, 256, 256, 128);
  k_gemm_wmma<128><<<dim3((NE + 127) / 128, 128 / 128), 256, 0, stream>>>(EMB, W16, Y32, NE, 128, 256);
  bn_stats(Y32, NE, 128);
  k_bn_act<<<blk((size_t)NE * 128), 256, 0, stream>>>(Y32, MEAN, ISTD, mg[1], mbe[1],
                                                      nullptr, EMB, (size_t)NE, 128, /*ReLU*/1);

  // L2: [E,128] x [128,64] -> Y32 ; bias + ReLU (no BN)
  k_cast_pad_f16<<<blk((size_t)128 * 64), 256, 0, stream>>>(mw2, W16, 128, 128, 64);
  k_gemm_wmma<64><<<dim3((NE + 255) / 256, 64 / 64), 256, 0, stream>>>(EMB, W16, Y32, NE, 64, 128);
  k_bias_relu<<<blk((size_t)NE * 64), 256, 0, stream>>>(Y32, mb2, (size_t)NE, 64);

  // L3: 64-wide dot + mb3 -> out[E]
  k_final_dot<<<blk((size_t)NE), 256, 0, stream>>>(Y32, mw3, mb3, out);
}
